// MDNSeqModel_49503793053905
// MI455X (gfx1250) — matrix operand order repaired and checked by reading, compiled, not run on hardware
//
#include <hip/hip_runtime.h>
#include <hip/hip_bf16.h>

// ---------------------------------------------------------------------------
// MDN sequential model on gfx1250 (CDNA5): bf16 WMMA GEMMs, persistent scan.
// v4b: weight loads via address_space(1) pointers (global_load, LOADcnt only);
//      native ext_vector 16B load type (HIP's uint4 class can't cross AS);
//      wide load clauses before WMMA blocks; dual accumulators in helpers.
// ---------------------------------------------------------------------------

#define STATE 21
#define ACTD  8
#define ZD    128
#define HD    512
#define BD    1024
#define TD    128
#define BT    32      // batch tile per workgroup
#define NTH   1024    // threads per block (32 wave32)
#define NWG   (BD / BT)

typedef __bf16 bf16;
typedef __attribute__((address_space(1))) bf16 gbf16;          // global bf16
typedef __attribute__((ext_vector_type(4))) unsigned int v4u;  // 16B chunk
typedef __attribute__((address_space(1))) v4u gv4u;            // global 16B
typedef __attribute__((ext_vector_type(16))) __bf16 v16bf;
typedef __attribute__((ext_vector_type(8)))  float  v8f;

// ---- workspace layout (bf16 elements) -------------------------------------
constexpr size_t OFF_WIH  = 0;                      // [2048][160]
constexpr size_t OFF_WHH  = OFF_WIH  + 2048 * 160;  // [2048][512]
constexpr size_t OFF_WE1  = OFF_WHH  + 2048 * 512;  // [256][512]
constexpr size_t OFF_WE2  = OFF_WE1  + 256 * 512;   // [128][256]
constexpr size_t OFF_WEZ  = OFF_WE2  + 128 * 256;   // [256][128]
constexpr size_t OFF_WD1  = OFF_WEZ  + 256 * 128;   // [64][160]
constexpr size_t OFF_WD2  = OFF_WD1  + 64 * 160;    // [64][64]
constexpr size_t OFF_WD3  = OFF_WD2  + 64 * 64;     // [32][64]
constexpr size_t OFF_WD4  = OFF_WD3  + 32 * 64;     // [16][32]
constexpr size_t OFF_WOUT = OFF_WD4  + 16 * 32;     // [64][32] (mu rows 0-20, sig rows 32-52)
constexpr size_t WS_BF16_ELEMS = OFF_WOUT + 64 * 32;
constexpr size_t OFF_ACC_BYTES = ((WS_BF16_ELEMS * 2 + 255) / 256) * 256;
constexpr size_t WS_NEEDED     = OFF_ACC_BYTES + 16;

// dynamic LDS bytes: bf16 activations (incl. double-buffered h) + f32 buffers
constexpr size_t SMEM_BYTES =
    (size_t)BT * (160 + 512 + 512 + 256 + 128 + 128 + 160 + 64 + 64 + 32 + 32) * 2 +
    (size_t)BT * (256 + 64) * 4;   // = 172032

// ---- WMMA helpers ----------------------------------------------------------
__device__ __forceinline__ v8f zero8() {
  v8f z = {0.f, 0.f, 0.f, 0.f, 0.f, 0.f, 0.f, 0.f};
  return z;
}
__device__ __forceinline__ v8f wmma_bf(v16bf a, v16bf b, v8f c) {
  return __builtin_amdgcn_wmma_f32_16x16x32_bf16(false, a, false, b, (short)0, c,
                                                 false, false);
}
// A fragment 16x32 (MxK) from LDS, row-major; lane half h holds k = 16h..16h+15 (same order as frB)
__device__ __forceinline__ v16bf frA(const bf16* base, int ld, int row0, int k0) {
  const int ln = threadIdx.x & 31;
  const bf16* p = base + (size_t)(row0 + (ln & 15)) * ld + k0 + ((ln >> 4) << 4);
  union {
    v4u q[2];
    v16bf v;
  } u;
  u.q[0] = *reinterpret_cast<const v4u*>(p);
  u.q[1] = *reinterpret_cast<const v4u*>(p + 8);
  return u.v;
}
// B fragment 32x16 (KxN) from row-major W[N][K] in GLOBAL memory
__device__ __forceinline__ v16bf frB(const gbf16* base, int ld, int n0, int k0) {
  const int ln = threadIdx.x & 31;
  const gbf16* p = base + (size_t)(n0 + (ln & 15)) * ld + k0 + ((ln >> 4) << 4);
  union {
    v4u q[2];
    v16bf v;
  } u;
  u.q[0] = *(const gv4u*)p;
  u.q[1] = *(const gv4u*)(p + 8);
  return u.v;
}
// defeat LICM (which previously spilled every weight fragment to scratch) and
// pin the pointer to the GLOBAL address space so loads are global_load, not flat
__device__ __forceinline__ const gbf16* launder(const bf16* p) {
  unsigned long long u = (unsigned long long)p;
  asm volatile("" : "+s"(u));
  return (const gbf16*)u;
}

// one 16x16 C tile per wave: rows [row0,row0+16), cols [nb,nb+16)
// dual accumulators break the WMMA->WMMA dependency chain
__device__ __forceinline__ v8f tile1_acc(const bf16* A, int lda, const gbf16* W,
                                         int ldw, int nb, int row0, int ksteps) {
  v8f c0 = zero8(), c1 = zero8();
  int ks = 0;
  for (; ks + 1 < ksteps; ks += 2) {
    v16bf a0 = frA(A, lda, row0, ks * 32);
    v16bf b0 = frB(W, ldw, nb, ks * 32);
    v16bf a1 = frA(A, lda, row0, ks * 32 + 32);
    v16bf b1 = frB(W, ldw, nb, ks * 32 + 32);
    c0 = wmma_bf(a0, b0, c0);
    c1 = wmma_bf(a1, b1, c1);
  }
  for (; ks < ksteps; ++ks) {
    v16bf a = frA(A, lda, row0, ks * 32);
    v16bf b = frB(W, ldw, nb, ks * 32);
    c0 = wmma_bf(a, b, c0);
  }
  return c0 + c1;
}
__device__ __forceinline__ void tile1_bf(const bf16* A, int lda, const gbf16* W,
                                         int ldw, int nb, int row0, int ksteps,
                                         float bias, bf16* D, int ldd, bool relu) {
  const int ln = threadIdx.x & 31, l15 = ln & 15, lhi = ln >> 4;
  v8f c = tile1_acc(A, lda, W, ldw, nb, row0, ksteps);
  const int col = nb + l15;
#pragma unroll
  for (int v = 0; v < 8; ++v) {
    float x = c[v] + bias;
    if (relu) x = fmaxf(x, 0.f);
    D[(row0 + lhi * 8 + v) * ldd + col] = (bf16)x;
  }
}
__device__ __forceinline__ void tile1_f32(const bf16* A, int lda, const gbf16* W,
                                          int ldw, int nb, int row0, int ksteps,
                                          float bias, float* D, int ldd) {
  const int ln = threadIdx.x & 31, l15 = ln & 15, lhi = ln >> 4;
  v8f c = tile1_acc(A, lda, W, ldw, nb, row0, ksteps);
  const int col = nb + l15;
#pragma unroll
  for (int v = 0; v < 8; ++v) D[(row0 + lhi * 8 + v) * ldd + col] = c[v] + bias;
}

// ---- deterministic stand-in for jax.random.normal eps ----------------------
__device__ __forceinline__ unsigned hash_u32(unsigned x) {
  x ^= x >> 16;
  x *= 0x7feb352dU;
  x ^= x >> 15;
  x *= 0x846ca68bU;
  x ^= x >> 16;
  return x;
}
__device__ __forceinline__ float eps_normal(int t, int b, int z) {
  unsigned s = ((unsigned)(t * 1031 + b) << 8) ^ (unsigned)z;
  unsigned u1 = hash_u32(s * 0x9E3779B1U + 0xA511E9B3U);
  unsigned u2 = hash_u32(s * 0x85EBCA77U + 0x63D83595U);
  float f1 = (float)(u1 >> 8) * (1.0f / 16777216.0f) + 1e-7f;
  float f2 = (float)(u2 >> 8) * (1.0f / 16777216.0f);
  return sqrtf(-2.0f * __logf(f1)) * __cosf(6.28318530718f * f2);
}

// ---- setup kernels ----------------------------------------------------------
__global__ void k_zero_f(float* p, int n) {
  int i = blockIdx.x * blockDim.x + threadIdx.x;
  if (i < n) p[i] = 0.f;
}
__global__ void k_zero_bf(bf16* p, int n) {
  int i = blockIdx.x * blockDim.x + threadIdx.x;
  if (i < n) p[i] = (bf16)0.f;
}
__global__ void k_convert_pad(const float* __restrict__ src, bf16* __restrict__ dst,
                              int rows, int K, int Kpad) {
  int i = blockIdx.x * blockDim.x + threadIdx.x;
  int total = rows * Kpad;
  if (i >= total) return;
  int r = i / Kpad, k = i - r * Kpad;
  float v = (k < K) ? src[(size_t)r * K + k] : 0.f;
  dst[i] = (bf16)v;
}
__global__ void k_finalize(const float* __restrict__ acc, float* __restrict__ out) {
  out[0] = acc[0] / (float)(BD * TD);
  out[1] = acc[1] / (float)((size_t)BD * TD * STATE);
  out[2] = acc[2] / (float)((size_t)BD * TD * STATE);
}

// ---- main persistent scan kernel -------------------------------------------
__global__ __launch_bounds__(NTH) void mdn_scan_kernel(
    const float* __restrict__ Aact, const float* __restrict__ X,
    const float* __restrict__ Y, const float* __restrict__ b_ih,
    const float* __restrict__ b_hh, const float* __restrict__ enc_b1,
    const float* __restrict__ enc_b2, const float* __restrict__ enc_bz,
    const float* __restrict__ dec_b1, const float* __restrict__ dec_b2,
    const float* __restrict__ dec_b3, const float* __restrict__ dec_b4,
    const float* __restrict__ mu_b, const float* __restrict__ sig_b,
    const bf16* __restrict__ Wih, const bf16* __restrict__ Whh,
    const bf16* __restrict__ We1, const bf16* __restrict__ We2,
    const bf16* __restrict__ Wez, const bf16* __restrict__ Wd1,
    const bf16* __restrict__ Wd2, const bf16* __restrict__ Wd3,
    const bf16* __restrict__ Wd4, const bf16* __restrict__ Wout,
    float* __restrict__ accG) {
  extern __shared__ unsigned char smem[];
  bf16* s_rnn = (bf16*)smem;               // [32][160]  rnn_in (a|z|pad)
  bf16* s_hA  = s_rnn + BT * 160;          // [32][512]  h ping
  bf16* s_hB  = s_hA  + BT * 512;          // [32][512]  h pong
  bf16* s_e1  = s_hB  + BT * 512;          // [32][256]
  bf16* s_e2  = s_e1  + BT * 256;          // [32][128]
  bf16* s_z   = s_e2  + BT * 128;          // [32][128]
  bf16* s_dec = s_z   + BT * 128;          // [32][160]  (obs|z|pad)
  bf16* s_d1  = s_dec + BT * 160;          // [32][64]
  bf16* s_d2  = s_d1  + BT * 64;           // [32][64]
  bf16* s_d3  = s_d2  + BT * 64;           // [32][32]
  bf16* s_d4  = s_d3  + BT * 32;           // [32][32] (cols 16-31 stay zero)
  float* s_zz  = (float*)(s_d4 + BT * 32); // [32][256]
  float* s_out = s_zz + BT * 256;          // [32][64]

  const int tid = threadIdx.x;
  const int wv = tid >> 5, ln = tid & 31, l15 = ln & 15, lhi = ln >> 4;
  const int bG0 = blockIdx.x * BT;

  // gate-phase mapping: wave owns (batch half, 32-wide hidden slice)
  const int mtg = (wv >> 4) * 16;   // row0: 0 or 16
  const int jt = (wv & 15) * 32;    // hidden slice base
  // generic per-phase job mapping: job nt = wv>>1, row0 = (wv&1)*16
  const int jnt = wv >> 1;
  const int jr0 = (wv & 1) * 16;

  // ---- one-time init ----
  for (int i = tid; i < BT * HD; i += NTH) {
    s_hA[i] = (bf16)0.f;
    s_hB[i] = (bf16)0.f;
  }
  for (int i = tid; i < BT * ZD; i += NTH) s_z[i] = (bf16)0.f;
  for (int i = tid; i < BT * 160; i += NTH) s_rnn[i] = (bf16)0.f;
  for (int i = tid; i < BT * 160; i += NTH) s_dec[i] = (bf16)0.f;
  for (int i = tid; i < BT * 32; i += NTH) s_d4[i] = (bf16)0.f;
  for (int i = tid; i < BT * STATE; i += NTH) {
    int b = i / STATE, s = i - b * STATE;
    s_dec[b * 160 + s] = (bf16)X[((size_t)(bG0 + b) * TD) * STATE + s]; // obs = x[:,0]
  }

  // per-lane bias precompute
  float bsum[4][2];
#pragma unroll
  for (int g = 0; g < 4; ++g)
#pragma unroll
    for (int p = 0; p < 2; ++p) {
      int j = g * HD + jt + p * 16 + l15;
      bsum[g][p] = b_ih[j] + b_hh[j];
    }
  float bE1 = enc_b1[jnt * 16 + l15];
  float bE2 = (wv < 16) ? enc_b2[jnt * 16 + l15] : 0.f;
  float bEz = enc_bz[jnt * 16 + l15];
  float bD1 = (wv < 8) ? dec_b1[jnt * 16 + l15] : 0.f;
  float bD2 = (wv < 8) ? dec_b2[jnt * 16 + l15] : 0.f;
  float bD3 = (wv < 4) ? dec_b3[jnt * 16 + l15] : 0.f;
  float bD4 = (wv < 2) ? dec_b4[l15] : 0.f;

  float creg[2][8];
#pragma unroll
  for (int p = 0; p < 2; ++p)
#pragma unroll
    for (int v = 0; v < 8; ++v) creg[p][v] = 0.f;

  float aLog = 0.f, aMae = 0.f, aSd = 0.f;
  __syncthreads();

  for (int t = 0; t < TD; ++t) {
    // opaque, global-addrspace copies of the weight pointers (per iteration)
    const gbf16* Wih_t = launder(Wih);
    const gbf16* Whh_t = launder(Whh);
    const gbf16* We1_t = launder(We1);
    const gbf16* We2_t = launder(We2);
    const gbf16* Wez_t = launder(Wez);
    const gbf16* Wd1_t = launder(Wd1);
    const gbf16* Wd2_t = launder(Wd2);
    const gbf16* Wd3_t = launder(Wd3);
    const gbf16* Wd4_t = launder(Wd4);
    const gbf16* Wout_t = launder(Wout);

    const bf16* hc = (t & 1) ? s_hB : s_hA;  // h input (previous step)
    bf16* hn = (t & 1) ? s_hA : s_hB;        // h output (this step)

    // -- 1. build rnn_in = [a_t | z_prev | 0pad] --
    for (int i = tid; i < BT * 136; i += NTH) {
      int b = i / 136, k = i - b * 136;
      float v = (k < ACTD) ? Aact[((size_t)(bG0 + b) * TD + t) * ACTD + k]
                           : (float)s_z[b * ZD + (k - ACTD)];
      s_rnn[b * 160 + k] = (bf16)v;
    }
    __syncthreads();

    // -- 2. LSTM gates: two passes of 16 hidden cols, 4 accumulators each --
#pragma unroll
    for (int p = 0; p < 2; ++p) {
      const int jp = jt + p * 16;
      v8f acc[4];
#pragma unroll
      for (int i = 0; i < 4; ++i) acc[i] = zero8();
      for (int ks = 0; ks < 5; ++ks) {  // rnn_in @ W_ih^T, K=160
        // issue all 5 loads as one clause, then the 4 WMMAs
        v16bf a = frA(s_rnn, 160, mtg, ks * 32);
        v16bf b0 = frB(Wih_t, 160, 0 * HD + jp, ks * 32);
        v16bf b1 = frB(Wih_t, 160, 1 * HD + jp, ks * 32);
        v16bf b2 = frB(Wih_t, 160, 2 * HD + jp, ks * 32);
        v16bf b3 = frB(Wih_t, 160, 3 * HD + jp, ks * 32);
        acc[0] = wmma_bf(a, b0, acc[0]);
        acc[1] = wmma_bf(a, b1, acc[1]);
        acc[2] = wmma_bf(a, b2, acc[2]);
        acc[3] = wmma_bf(a, b3, acc[3]);
      }
      for (int ks = 0; ks < 16; ++ks) {  // h @ W_hh^T, K=512
        v16bf a = frA(hc, HD, mtg, ks * 32);
        v16bf b0 = frB(Whh_t, HD, 0 * HD + jp, ks * 32);
        v16bf b1 = frB(Whh_t, HD, 1 * HD + jp, ks * 32);
        v16bf b2 = frB(Whh_t, HD, 2 * HD + jp, ks * 32);
        v16bf b3 = frB(Whh_t, HD, 3 * HD + jp, ks * 32);
        acc[0] = wmma_bf(a, b0, acc[0]);
        acc[1] = wmma_bf(a, b1, acc[1]);
        acc[2] = wmma_bf(a, b2, acc[2]);
        acc[3] = wmma_bf(a, b3, acc[3]);
      }
      // cell update in registers, write new h into the other buffer
#pragma unroll
      for (int v = 0; v < 8; ++v) {
        const int row = mtg + lhi * 8 + v;
        float iv = acc[0][v] + bsum[0][p];
        float fv = acc[1][v] + bsum[1][p];
        float gv = acc[2][v] + bsum[2][p];
        float ov = acc[3][v] + bsum[3][p];
        iv = 1.f / (1.f + __expf(-iv));
        fv = 1.f / (1.f + __expf(-fv));
        ov = 1.f / (1.f + __expf(-ov));
        gv = tanhf(gv);
        float c = fv * creg[p][v] + iv * gv;
        creg[p][v] = c;
        hn[row * HD + jp + l15] = (bf16)(ov * tanhf(c));
      }
    }
    __syncthreads();

    // -- 3. encoder layer 1: [32,512]x[512,256] relu (32 tile jobs) --
    tile1_bf(hn, HD, We1_t, HD, jnt * 16, jr0, 16, bE1, s_e1, 256, true);
    __syncthreads();
    // -- 4. encoder layer 2: [32,256]x[256,128] relu (16 jobs) --
    if (wv < 16) tile1_bf(s_e1, 256, We2_t, 256, jnt * 16, jr0, 8, bE2, s_e2, 128, true);
    __syncthreads();
    // -- 5. zz: [32,128]x[128,256] (32 jobs) --
    tile1_f32(s_e2, 128, Wez_t, 128, jnt * 16, jr0, 4, bEz, s_zz, 256);
    __syncthreads();
    // -- 6. z = z_loc + exp(1+z_sig)*eps ; write z and dec_in --
    for (int i = tid; i < BT * ZD; i += NTH) {
      int b = i >> 7, zi = i & 127;
      float zl = s_zz[b * 256 + zi];
      float zs = fmaxf(__expf(1.0f + s_zz[b * 256 + ZD + zi]), 1e-20f);
      float zval = zl + zs * eps_normal(t, bG0 + b, zi);
      bf16 zb = (bf16)zval;
      s_z[b * ZD + zi] = zb;                  // carried into next rnn_in
      s_dec[b * 160 + STATE + zi] = zb;       // decoder input
    }
    __syncthreads();
    // -- 7..10. decoder MLP --
    if (wv < 8) tile1_bf(s_dec, 160, Wd1_t, 160, jnt * 16, jr0, 5, bD1, s_d1, 64, true);
    __syncthreads();
    if (wv < 8) tile1_bf(s_d1, 64, Wd2_t, 64, jnt * 16, jr0, 2, bD2, s_d2, 64, true);
    __syncthreads();
    if (wv < 4) tile1_bf(s_d2, 64, Wd3_t, 64, jnt * 16, jr0, 2, bD3, s_d3, 32, true);
    __syncthreads();
    if (wv < 2) tile1_bf(s_d3, 32, Wd4_t, 32, 0, jr0, 1, bD4, s_d4, 32, true);
    __syncthreads();
    // -- 11. mu/sigma heads (packed [64][32] weight, K padded to 32) --
    if (wv < 8) tile1_f32(s_d4, 32, Wout_t, 32, jnt * 16, jr0, 1, 0.f, s_out, 64);
    __syncthreads();
    // -- 12. log-likelihood / mae / sd accumulation --
    for (int i = tid; i < BT * STATE; i += NTH) {
      int b = i / STATE, s = i - b * STATE;
      float mu = s_out[b * 64 + s] + mu_b[s];
      float sg = fmaxf(__expf(1.0f + s_out[b * 64 + 32 + s] + sig_b[s]), 1e-20f);
      float yv = Y[((size_t)(bG0 + b) * TD + t) * STATE + s];
      float r = (yv - mu) / sg;
      aLog += 0.5f * r * r + __logf(sg) + 0.91893853320467274f;
      aMae += fabsf(mu - yv);
      aSd += sg;
    }
    __syncthreads();
  }

  atomicAdd(&accG[0], aLog);
  atomicAdd(&accG[1], aMae);
  atomicAdd(&accG[2], aSd);
}

// ---------------------------------------------------------------------------
extern "C" void kernel_launch(void* const* d_in, const int* in_sizes, int n_in,
                              void* d_out, int out_size, void* d_ws, size_t ws_size,
                              hipStream_t stream) {
  if (ws_size < WS_NEEDED) return;
  const float* X      = (const float*)d_in[0];
  const float* Aact   = (const float*)d_in[1];
  const float* Y      = (const float*)d_in[2];
  const float* W_ih   = (const float*)d_in[3];
  const float* W_hh   = (const float*)d_in[4];
  const float* b_ih   = (const float*)d_in[5];
  const float* b_hh   = (const float*)d_in[6];
  const float* enc_w1 = (const float*)d_in[7];
  const float* enc_b1 = (const float*)d_in[8];
  const float* enc_w2 = (const float*)d_in[9];
  const float* enc_b2 = (const float*)d_in[10];
  const float* enc_wz = (const float*)d_in[11];
  const float* enc_bz = (const float*)d_in[12];
  const float* dec_w1 = (const float*)d_in[13];
  const float* dec_b1 = (const float*)d_in[14];
  const float* dec_w2 = (const float*)d_in[15];
  const float* dec_b2 = (const float*)d_in[16];
  const float* dec_w3 = (const float*)d_in[17];
  const float* dec_b3 = (const float*)d_in[18];
  const float* dec_w4 = (const float*)d_in[19];
  const float* dec_b4 = (const float*)d_in[20];
  const float* mu_w   = (const float*)d_in[21];
  const float* mu_b   = (const float*)d_in[22];
  const float* sig_w  = (const float*)d_in[23];
  const float* sig_b  = (const float*)d_in[24];

  bf16* wsb = (bf16*)d_ws;
  float* accG = (float*)((char*)d_ws + OFF_ACC_BYTES);

  k_zero_f<<<1, 64, 0, stream>>>(accG, 3);
  k_zero_bf<<<(64 * 32 + 255) / 256, 256, 0, stream>>>(wsb + OFF_WOUT, 64 * 32);

  auto conv = [&](const float* s, size_t off, int r, int K, int Kp) {
    int tot = r * Kp;
    k_convert_pad<<<(tot + 255) / 256, 256, 0, stream>>>(s, wsb + off, r, K, Kp);
  };
  conv(W_ih, OFF_WIH, 2048, 136, 160);
  conv(W_hh, OFF_WHH, 2048, 512, 512);
  conv(enc_w1, OFF_WE1, 256, 512, 512);
  conv(enc_w2, OFF_WE2, 128, 256, 256);
  conv(enc_wz, OFF_WEZ, 256, 128, 128);
  conv(dec_w1, OFF_WD1, 64, 149, 160);
  conv(dec_w2, OFF_WD2, 64, 64, 64);
  conv(dec_w3, OFF_WD3, 32, 64, 64);
  conv(dec_w4, OFF_WD4, 16, 32, 32);
  conv(mu_w, OFF_WOUT, 21, 16, 32);               // rows 0-20
  conv(sig_w, OFF_WOUT + 32 * 32, 21, 16, 32);    // rows 32-52

  mdn_scan_kernel<<<NWG, NTH, SMEM_BYTES, stream>>>(
      Aact, X, Y, b_ih, b_hh, enc_b1, enc_b2, enc_bz, dec_b1, dec_b2, dec_b3,
      dec_b4, mu_b, sig_b, wsb + OFF_WIH, wsb + OFF_WHH, wsb + OFF_WE1,
      wsb + OFF_WE2, wsb + OFF_WEZ, wsb + OFF_WD1, wsb + OFF_WD2, wsb + OFF_WD3,
      wsb + OFF_WD4, wsb + OFF_WOUT, accG);

  k_finalize<<<1, 1, 0, stream>>>(accG, (float*)d_out);
}
